// RPN_34153579938445
// MI455X (gfx1250) — matrix operand order repaired and checked by reading, not compiled, not run on hardware
//
#include <hip/hip_runtime.h>
#include <hip/hip_bf16.h>
#include <cmath>

typedef __attribute__((ext_vector_type(16))) _Float16 v16h;
typedef __attribute__((ext_vector_type(8)))  float    v8f;

struct BaseA { float v[12]; };

// ---------------------------------------------------------------------------
// 1a) convert conv weights f32 -> f16 with K reordered to (tap p, channel c):
//     wf16[m][p*256 + c] = conv_w[m][c*9 + p]   (A matrix [256][2304])
// ---------------------------------------------------------------------------
__global__ void k_cvt(const float* __restrict__ w, _Float16* __restrict__ o, int n) {
    int i = blockIdx.x * 256 + threadIdx.x;
    if (i >= n) return;
    int m = i / 2304, rem = i % 2304;
    int p = rem >> 8, c = rem & 255;
    o[i] = (_Float16)w[m * 2304 + c * 9 + p];
}

// ---------------------------------------------------------------------------
// 1b) per level: features f32 [256][H][W] -> padded transposed f16
//     featT[(y+1)*(W+2) + (x+1)][256], border = 0 (physical conv padding)
// ---------------------------------------------------------------------------
__global__ void k_cvtf(const float* __restrict__ feat, _Float16* __restrict__ featT,
                       int H, int W, int HW)
{
    int Wp = W + 2, Hp = H + 2;
    int total = Hp * Wp * 256;
    int i = blockIdx.x * 256 + threadIdx.x;
    if (i >= total) return;
    int c = i & 255;
    int sp = i >> 8;
    int xx = sp % Wp, yy = sp / Wp;
    float v = 0.f;
    int x = xx - 1, y = yy - 1;
    if ((unsigned)x < (unsigned)W && (unsigned)y < (unsigned)H)
        v = feat[(size_t)c * HW + y * W + x];
    featT[i] = (_Float16)v;
}

// ---------------------------------------------------------------------------
// 2) 3x3 conv (pad 1) as implicit GEMM, WMMA f16->f32, fused bias+ReLU.
//    Workgroup: 256 threads = 8 waves; block tile = 32 outC x 128 spatial;
//    each wave computes 16M x 32N (two accumulators, A-fragment reuse).
//    K = 2304 = 9 taps x 256 channels, stepped 64/iter (4 WMMA per wave).
//    A/B tiles staged with GLOBAL_LOAD_ASYNC_TO_LDS_B128, double-buffered.
// ---------------------------------------------------------------------------
__global__ __launch_bounds__(256) void k_conv(
    const _Float16* __restrict__ featT, const _Float16* __restrict__ wf,
    const float* __restrict__ bias, _Float16* __restrict__ t,
    int H, int W, int HW)
{
    __shared__ unsigned lA[2][32 * 32];    // 32 M x 64 K halfs (packed pairs)  8KB
    __shared__ unsigned lB[2][128 * 32];   // 128 N x 64 K halfs (packed pairs) 32KB

    const int tid  = threadIdx.x;
    const int lane = tid & 31;
    const int wv   = tid >> 5;
    const int m0   = blockIdx.y * 32;
    const int n0   = blockIdx.x * 128;
    const int Wp   = W + 2;

    // staging assignments
    const int am = tid >> 3;          // A row (M) 0..31
    const int ac = tid & 7;           // A 16B chunk within K64
    const int bn = tid & 127;         // B col (N) 0..127
    const int bq = tid >> 7;          // B half 0..1 (4 chunks each)
    int ng = n0 + bn; if (ng >= HW) ng = HW - 1;   // clamp: dead cols never stored
    const int by = ng / W, bx = ng % W;

    // wave tile coords: 2 M-subtiles x 4 N-subtiles (each wave 16M x 32N)
    const int mi = wv >> 2, ni = wv & 3;
    const int mrow = lane & 15, ncol = lane & 15;
    const int hh = (lane >= 16) ? 1 : 0;
    const int mloc  = mi * 16 + mrow;
    const int nloc0 = ni * 32 + ncol;      // first 16x16 subtile cols
    const int nloc1 = nloc0 + 16;          // second 16x16 subtile cols

    v8f acc0 = {}, acc1 = {};

    auto stage = [&](int blk, int buf) {
        const int k0 = blk * 64;
        const int p  = k0 >> 8;                // 3x3 tap
        const int cb = k0 & 255;               // channel base
        const int ky = p / 3, kx = p - 3 * ky; // uniform per block
        // A: 32x64 halfs = 4KB -> 1 async b128 per lane
        {
            const _Float16* ga = wf + (size_t)(m0 + am) * 2304 + k0 + ac * 8;
            unsigned la = (unsigned)(size_t)&lA[buf][am * 32 + ac * 4];
            asm volatile("global_load_async_to_lds_b128 %0, %1, off"
                         :: "v"(la), "v"(ga) : "memory");
        }
        // B: 128x64 halfs = 16KB -> 4 async b128 per lane (contiguous channels)
        {
            const int sp = (by + ky) * Wp + (bx + kx);
            const _Float16* gb = featT + (size_t)sp * 256 + cb + bq * 32;
            unsigned lb = (unsigned)(size_t)&lB[buf][bn * 32 + bq * 16];
            #pragma unroll
            for (int j = 0; j < 4; ++j) {
                const _Float16* gj = gb + j * 8;
                unsigned lj = lb + j * 16;
                asm volatile("global_load_async_to_lds_b128 %0, %1, off"
                             :: "v"(lj), "v"(gj) : "memory");
            }
        }
    };

    stage(0, 0);
    for (int blk = 0; blk < 36; ++blk) {
        const int buf = blk & 1;
        if (blk < 35) {
            stage(blk + 1, buf ^ 1);
            // 5 outstanding = next group; this group complete (in-order).
            asm volatile("s_wait_asynccnt 0x5" ::: "memory");
        } else {
            asm volatile("s_wait_asynccnt 0x0" ::: "memory");
        }
        __syncthreads();
        #pragma unroll
        for (int s = 0; s < 2; ++s) {
            v16h afr, bfr0, bfr1;
            #pragma unroll
            for (int v = 0; v < 8; ++v) {
                // A: lanes0-15 K pairs {0..7,16..23}; lanes16-31 shifted by 8
                int ia = mloc * 32 + s * 16 + ((v < 4) ? v : v + 4) + hh * 4;
                union { unsigned u; _Float16 x[2]; } ua; ua.u = lA[buf][ia];
                afr[2 * v] = ua.x[0]; afr[2 * v + 1] = ua.x[1];
                // B: lanes0-15 K pairs {0..7,16..23}; lanes16-31 shifted by 8
                int ib0 = nloc0 * 32 + s * 16 + ((v < 4) ? v : v + 4) + hh * 4;
                union { unsigned u; _Float16 x[2]; } ub0; ub0.u = lB[buf][ib0];
                bfr0[2 * v] = ub0.x[0]; bfr0[2 * v + 1] = ub0.x[1];
                int ib1 = nloc1 * 32 + s * 16 + ((v < 4) ? v : v + 4) + hh * 4;
                union { unsigned u; _Float16 x[2]; } ub1; ub1.u = lB[buf][ib1];
                bfr1[2 * v] = ub1.x[0]; bfr1[2 * v + 1] = ub1.x[1];
            }
            acc0 = __builtin_amdgcn_wmma_f32_16x16x32_f16(
                false, afr, false, bfr0, (short)0, acc0, false, false);
            acc1 = __builtin_amdgcn_wmma_f32_16x16x32_f16(
                false, afr, false, bfr1, (short)0, acc1, false, false);
        }
        __syncthreads();
    }

    // epilogue: bias + ReLU, write t[m][n] as f16
    {
        const int mw  = m0 + mi * 16;
        const int nn0 = n0 + nloc0;
        const int nn1 = n0 + nloc1;
        #pragma unroll
        for (int r2 = 0; r2 < 8; ++r2) {
            int m = mw + r2 + (hh ? 8 : 0);
            float b = bias[m];
            if (nn0 < HW) {
                float val = acc0[r2] + b;
                val = val > 0.f ? val : 0.f;
                t[(size_t)m * HW + nn0] = (_Float16)val;
            }
            if (nn1 < HW) {
                float val = acc1[r2] + b;
                val = val > 0.f ? val : 0.f;
                t[(size_t)m * HW + nn1] = (_Float16)val;
            }
        }
    }
}

// ---------------------------------------------------------------------------
// 3) 1x1 heads: cls (3) + bbox (12) logits per pixel; weights in LDS.
// ---------------------------------------------------------------------------
__global__ __launch_bounds__(256) void k_heads(
    const _Float16* __restrict__ t,
    const float* __restrict__ clsw, const float* __restrict__ clsb,
    const float* __restrict__ bxw,  const float* __restrict__ bxb,
    float* __restrict__ s, float* __restrict__ d, int HW)
{
    __shared__ float lw[15 * 256];
    for (int i = threadIdx.x; i < 3 * 256; i += 256)  lw[i] = clsw[i];
    for (int i = threadIdx.x; i < 12 * 256; i += 256) lw[768 + i] = bxw[i];
    __syncthreads();

    int n = blockIdx.x * 256 + threadIdx.x;
    if (n >= HW) return;

    float acc[15];
    #pragma unroll
    for (int o = 0; o < 15; ++o) acc[o] = 0.f;
    for (int c = 0; c < 256; ++c) {
        float tc = (float)t[(size_t)c * HW + n];
        #pragma unroll
        for (int o = 0; o < 15; ++o) acc[o] += lw[o * 256 + c] * tc;
    }
    #pragma unroll
    for (int a = 0; a < 3; ++a) s[n * 3 + a] = acc[a] + clsb[a];
    #pragma unroll
    for (int a = 0; a < 3; ++a)
        #pragma unroll
        for (int j = 0; j < 4; ++j)
            d[(size_t)(n * 3 + a) * 4 + j] = acc[3 + a * 4 + j] + bxb[a * 4 + j];
}

// ---------------------------------------------------------------------------
// 4) sort keys: (monotone(score) << 32) | (0xFFFFFFFF - idx); pad = 0.
// ---------------------------------------------------------------------------
__global__ void k_keys(const float* __restrict__ s, unsigned long long* __restrict__ keys,
                       int A, int P)
{
    int i = blockIdx.x * 256 + threadIdx.x;
    if (i >= P) return;
    unsigned long long key = 0ULL;
    if (i < A) {
        unsigned u = __float_as_uint(s[i]);
        u = (u & 0x80000000u) ? ~u : (u | 0x80000000u);
        key = ((unsigned long long)u << 32) |
              (unsigned long long)(0xFFFFFFFFu - (unsigned)i);
    }
    keys[i] = key;
}

__global__ void k_bstep(unsigned long long* __restrict__ keys, int j, int kk)
{
    int i = blockIdx.x * 256 + threadIdx.x;
    int ixj = i ^ j;
    if (ixj > i) {
        unsigned long long a = keys[i], b = keys[ixj];
        bool descRegion = ((i & kk) == 0);
        bool sw = descRegion ? (a < b) : (a > b);
        if (sw) { keys[i] = b; keys[ixj] = a; }
    }
}

// ---------------------------------------------------------------------------
// 5) per-level box decode from sorted keys (top-k entries)
// ---------------------------------------------------------------------------
__global__ void k_decode(const unsigned long long* __restrict__ keys,
    const float* __restrict__ s, const float* __restrict__ d,
    float* __restrict__ scores_all, int* __restrict__ valid_all,
    float* __restrict__ braw, float* __restrict__ boff,
    int W, int stride, int kcount, int lvl, int outoff, BaseA base)
{
    int r = blockIdx.x * 256 + threadIdx.x;
    if (r >= kcount) return;
    unsigned idx = 0xFFFFFFFFu - (unsigned)(keys[r] & 0xFFFFFFFFu);
    float score = s[idx];
    int a = idx % 3; int cell = idx / 3;
    int x = cell % W, y = cell / W;
    float sx = (float)(x * stride), sy = (float)(y * stride);
    float ax0 = sx + base.v[a * 4 + 0], ay0 = sy + base.v[a * 4 + 1];
    float ax1 = sx + base.v[a * 4 + 2], ay1 = sy + base.v[a * 4 + 3];
    float w_ = ax1 - ax0, h_ = ay1 - ay0;
    float cx = ax0 + 0.5f * w_, cy = ay0 + 0.5f * h_;
    float dx = d[(size_t)idx * 4 + 0], dy = d[(size_t)idx * 4 + 1];
    const float CL = 4.135166556742356f; // log(1000/16)
    float dw = fminf(d[(size_t)idx * 4 + 2], CL);
    float dh = fminf(d[(size_t)idx * 4 + 3], CL);
    float pcx = dx * w_ + cx, pcy = dy * h_ + cy;
    float pw = expf(dw) * w_;
    float ph = expf(dh) * h_;
    float x0 = pcx - 0.5f * pw, y0 = pcy - 0.5f * ph;
    float x1 = pcx + 0.5f * pw, y1 = pcy + 0.5f * ph;
    x0 = fminf(fmaxf(x0, 0.f), 800.f); x1 = fminf(fmaxf(x1, 0.f), 800.f);
    y0 = fminf(fmaxf(y0, 0.f), 800.f); y1 = fminf(fmaxf(y1, 0.f), 800.f);
    int valid = ((x1 - x0) >= 0.001f) && ((y1 - y0) >= 0.001f);
    float off = lvl * 801.0f;
    int pos = outoff + r;
    braw[pos * 4 + 0] = x0; braw[pos * 4 + 1] = y0;
    braw[pos * 4 + 2] = x1; braw[pos * 4 + 3] = y1;
    boff[pos * 4 + 0] = x0 + off; boff[pos * 4 + 1] = y0 + off;
    boff[pos * 4 + 2] = x1 + off; boff[pos * 4 + 3] = y1 + off;
    scores_all[pos] = score; valid_all[pos] = valid;
}

// ---------------------------------------------------------------------------
// 6) apply global sort order
// ---------------------------------------------------------------------------
__global__ void k_gather(const unsigned long long* __restrict__ gkeys,
    const float* __restrict__ scores_all, const int* __restrict__ valid_all,
    const float* __restrict__ braw, const float* __restrict__ boff,
    float* __restrict__ ss, int* __restrict__ sv,
    float* __restrict__ sraw, float* __restrict__ soff,
    float* __restrict__ sarea, int N)
{
    int r = blockIdx.x * 256 + threadIdx.x;
    if (r >= N) return;
    unsigned idx = 0xFFFFFFFFu - (unsigned)(gkeys[r] & 0xFFFFFFFFu);
    ss[r] = scores_all[idx]; sv[r] = valid_all[idx];
    float b0 = boff[idx * 4 + 0], b1 = boff[idx * 4 + 1];
    float b2 = boff[idx * 4 + 2], b3 = boff[idx * 4 + 3];
    soff[r * 4 + 0] = b0; soff[r * 4 + 1] = b1;
    soff[r * 4 + 2] = b2; soff[r * 4 + 3] = b3;
    sarea[r] = (b2 - b0) * (b3 - b1);
    sraw[r * 4 + 0] = braw[idx * 4 + 0]; sraw[r * 4 + 1] = braw[idx * 4 + 1];
    sraw[r * 4 + 2] = braw[idx * 4 + 2]; sraw[r * 4 + 3] = braw[idx * 4 + 3];
}

// ---------------------------------------------------------------------------
// 7) IoU bitmask
// ---------------------------------------------------------------------------
__global__ void k_iou(const float* __restrict__ soff, const float* __restrict__ sarea,
                      unsigned* __restrict__ mask, int N, int W32)
{
    int g = blockIdx.x * 256 + threadIdx.x;
    if (g >= N * W32) return;
    int i = g / W32, w = g % W32;
    float ix0 = soff[i * 4 + 0], iy0 = soff[i * 4 + 1];
    float ix1 = soff[i * 4 + 2], iy1 = soff[i * 4 + 3];
    float ia = sarea[i];
    unsigned bits = 0;
    int jbase = w * 32;
    for (int b = 0; b < 32; ++b) {
        int j = jbase + b;
        if (j < N && j > i) {
            float x0 = fmaxf(ix0, soff[j * 4 + 0]);
            float y0 = fmaxf(iy0, soff[j * 4 + 1]);
            float x1 = fminf(ix1, soff[j * 4 + 2]);
            float y1 = fminf(iy1, soff[j * 4 + 3]);
            float ww = fmaxf(x1 - x0, 0.f), hh2 = fmaxf(y1 - y0, 0.f);
            float inter = ww * hh2;
            float iou = inter / (ia + sarea[j] - inter);
            if (iou > 0.7f) bits |= (1u << b);
        }
    }
    mask[(size_t)i * W32 + w] = bits;
}

// ---------------------------------------------------------------------------
// 8) sequential NMS scan (single workgroup), outputs final_scores
// ---------------------------------------------------------------------------
__global__ __launch_bounds__(256) void k_nms(const unsigned* __restrict__ mask,
    const int* __restrict__ sv, const float* __restrict__ ss,
    float* __restrict__ fscore, int N, int W32)
{
    __shared__ unsigned kw[256];
    for (int w = threadIdx.x; w < W32; w += 256) {
        unsigned bits = 0;
        for (int b = 0; b < 32; ++b) {
            int j = w * 32 + b;
            if (j < N && sv[j]) bits |= (1u << b);
        }
        kw[w] = bits;
    }
    __syncthreads();
    for (int i = 0; i < N; ++i) {
        bool ki = (kw[i >> 5] >> (i & 31)) & 1u;
        __syncthreads();
        if (ki) {
            for (int w = threadIdx.x; w < W32; w += 256)
                kw[w] &= ~mask[(size_t)i * W32 + w];
        }
        __syncthreads();
    }
    for (int r = threadIdx.x; r < N; r += 256) {
        bool keep = (kw[r >> 5] >> (r & 31)) & 1u;
        fscore[r] = keep ? ss[r] : -__builtin_huge_valf();
    }
}

// ---------------------------------------------------------------------------
// 9) write final 1000 boxes
// ---------------------------------------------------------------------------
__global__ void k_out(const unsigned long long* __restrict__ fkeys,
                      const float* __restrict__ sraw, float* __restrict__ out)
{
    int r = blockIdx.x * 256 + threadIdx.x;
    if (r >= 1000) return;
    unsigned idx = 0xFFFFFFFFu - (unsigned)(fkeys[r] & 0xFFFFFFFFu);
    out[r * 4 + 0] = sraw[idx * 4 + 0];
    out[r * 4 + 1] = sraw[idx * 4 + 1];
    out[r * 4 + 2] = sraw[idx * 4 + 2];
    out[r * 4 + 3] = sraw[idx * 4 + 3];
}

// ---------------------------------------------------------------------------
extern "C" void kernel_launch(void* const* d_in, const int* in_sizes, int n_in,
                              void* d_out, int out_size, void* d_ws, size_t ws_size,
                              hipStream_t stream)
{
    const float* feat[5] = {
        (const float*)d_in[0], (const float*)d_in[1], (const float*)d_in[2],
        (const float*)d_in[3], (const float*)d_in[4] };
    const float* conv_w = (const float*)d_in[5];
    const float* conv_b = (const float*)d_in[6];
    const float* cls_w  = (const float*)d_in[7];
    const float* cls_b  = (const float*)d_in[8];
    const float* bbox_w = (const float*)d_in[9];
    const float* bbox_b = (const float*)d_in[10];

    static const int LHW5[5] = {200, 100, 50, 25, 13};
    static const int LSTR[5] = {4, 8, 16, 32, 61};
    static const int LOFF[5] = {0, 1000, 2000, 3000, 4000};
    const int N_ALL = 4507;
    const int W32 = (N_ALL + 31) / 32;   // 141
    const int PG = 8192;

    // ---- workspace carve-out
    char* ws = (char*)d_ws;
    size_t off = 0;
    auto carve = [&](size_t bytes) -> char* {
        char* p = ws + off;
        off += (bytes + 255) & ~(size_t)255;
        return p;
    };
    _Float16* wf16  = (_Float16*)carve((size_t)589824 * 2);
    _Float16* featT = (_Float16*)carve((size_t)202 * 202 * 256 * 2);
    _Float16* tbuf  = (_Float16*)carve((size_t)256 * 40000 * 2);
    float* s_l = (float*)carve((size_t)120000 * 4);
    float* d_l = (float*)carve((size_t)120000 * 16);
    unsigned long long* keys = (unsigned long long*)carve((size_t)131072 * 8);
    float* scores_all = (float*)carve((size_t)N_ALL * 4);
    int*   valid_all  = (int*)carve((size_t)N_ALL * 4);
    float* braw = (float*)carve((size_t)N_ALL * 16);
    float* boff = (float*)carve((size_t)N_ALL * 16);
    float* ss   = (float*)carve((size_t)N_ALL * 4);
    int*   sv   = (int*)carve((size_t)N_ALL * 4);
    float* sraw = (float*)carve((size_t)N_ALL * 16);
    float* soff = (float*)carve((size_t)N_ALL * 16);
    float* sarea = (float*)carve((size_t)N_ALL * 4);
    unsigned long long* gkeys = (unsigned long long*)carve((size_t)PG * 8);
    float* fscore = (float*)carve((size_t)N_ALL * 4);
    unsigned* maskb = (unsigned*)carve((size_t)N_ALL * W32 * 4);

    auto bitonic = [&](unsigned long long* kp, int P) {
        for (int kk = 2; kk <= P; kk <<= 1)
            for (int j = kk >> 1; j > 0; j >>= 1)
                k_bstep<<<P / 256, 256, 0, stream>>>(kp, j, kk);
    };

    // 1) weights to f16 (K reordered)
    k_cvt<<<589824 / 256, 256, 0, stream>>>(conv_w, wf16, 589824);

    static const double SIZESd[5] = {32., 64., 128., 256., 512.};
    static const double RATd[3] = {0.5, 1.0, 2.0};
    for (int l = 0; l < 5; ++l) {
        int H = LHW5[l], W = LHW5[l], HW = H * W;
        int Hp = H + 2, Wp = W + 2;
        int padN = Hp * Wp * 256;
        k_cvtf<<<(padN + 255) / 256, 256, 0, stream>>>(feat[l], featT, H, W, HW);
        int ntN = (HW + 127) / 128;
        k_conv<<<dim3(ntN, 8), 256, 0, stream>>>(featT, wf16, conv_b, tbuf, H, W, HW);
        k_heads<<<(HW + 255) / 256, 256, 0, stream>>>(tbuf, cls_w, cls_b,
                                                      bbox_w, bbox_b, s_l, d_l, HW);
        int A = HW * 3;
        int P = 256; while (P < A) P <<= 1;
        k_keys<<<P / 256, 256, 0, stream>>>(s_l, keys, A, P);
        bitonic(keys, P);
        int kc = (A < 1000) ? A : 1000;

        BaseA ba;
        for (int a = 0; a < 3; ++a) {
            double hr = std::sqrt(RATd[a]);
            double wr = 1.0 / hr;
            double wsz = wr * SIZESd[l], hsz = hr * SIZESd[l];
            ba.v[a * 4 + 0] = (float)std::nearbyint(-wsz / 2.0);
            ba.v[a * 4 + 1] = (float)std::nearbyint(-hsz / 2.0);
            ba.v[a * 4 + 2] = (float)std::nearbyint(wsz / 2.0);
            ba.v[a * 4 + 3] = (float)std::nearbyint(hsz / 2.0);
        }
        k_decode<<<(kc + 255) / 256, 256, 0, stream>>>(keys, s_l, d_l,
            scores_all, valid_all, braw, boff, W, LSTR[l], kc, l, LOFF[l], ba);
    }

    // global stable sort by score
    k_keys<<<PG / 256, 256, 0, stream>>>(scores_all, gkeys, N_ALL, PG);
    bitonic(gkeys, PG);
    k_gather<<<(N_ALL + 255) / 256, 256, 0, stream>>>(gkeys, scores_all, valid_all,
        braw, boff, ss, sv, sraw, soff, sarea, N_ALL);

    // IoU mask + NMS
    int nW = N_ALL * W32;
    k_iou<<<(nW + 255) / 256, 256, 0, stream>>>(soff, sarea, maskb, N_ALL, W32);
    k_nms<<<1, 256, 0, stream>>>(maskb, sv, ss, fscore, N_ALL, W32);

    // final top-1000 + output
    k_keys<<<PG / 256, 256, 0, stream>>>(fscore, gkeys, N_ALL, PG);
    bitonic(gkeys, PG);
    k_out<<<(1000 + 255) / 256, 256, 0, stream>>>(gkeys, sraw, (float*)d_out);
}
